// customConv2D_63642825392221
// MI455X (gfx1250) — matrix operand order repaired and checked by reading, not compiled, not run on hardware
//
#include <hip/hip_runtime.h>

typedef __attribute__((ext_vector_type(16))) _Float16 v16h;
typedef __attribute__((ext_vector_type(8)))  float    v8f;

#define N_IMG   32
#define C_IN    16
#define H_IMG   256
#define W_IMG   256
#define HW      (H_IMG * W_IMG)
#define K_OUT   32
#define RED     144   // 9 spatial taps * 16 channels, u = rs*16 + c ordering
#define RED_PAD 160   // padded to 5 chunks of 32
#define NCHUNK  5
#define TILE_W  16    // pixels per wave (M dimension)
#define WAVES   8     // waves per block (wave32)

// Compute one 16-pixel x 32-channel tile.  MASKED=false is the interior fast
// path: every tap is an immediate offset off a single per-lane base pointer,
// no bounds VALU at all.  MASKED=true clamps addresses and masks values.
template <bool MASKED>
__device__ __forceinline__ void conv_tile(
    const float* __restrict__ xlane,          // image+channel-block base
    const _Float16 (* __restrict__ wlds)[RED_PAD],
    int h, int wx, int l15, int half,
    v8f& acc0, v8f& acc1)
{
    const float* pbase = xlane + h * W_IMG + wx;   // lane's spatial origin

    #pragma unroll
    for (int chunk = 0; chunk < NCHUNK; ++chunk) {
        const int kkbase = chunk * 32;

        // A fragment (im2col): element idx -> K = kkbase + 8*half + idx (idx<8)
        // or kkbase + 16 + 8*half + (idx-8).  u = rs*16 + c ordering makes the
        // two 8-element groups 8 consecutive channels at compile-time taps
        // rs0 = 2*chunk, rs1 = 2*chunk+1.
        float tmp[16];
        bool  ok[2];

        #pragma unroll
        for (int g = 0; g < 2; ++g) {
            const int rs = chunk * 2 + g;
            if (rs < 9) {
                const int r = rs / 3;           // compile-time
                const int s = rs - r * 3;       // compile-time
                if constexpr (!MASKED) {
                    ok[g] = true;
                    const float* p = pbase + (r - 1) * W_IMG + (s - 1);
                    #pragma unroll
                    for (int j = 0; j < 8; ++j)
                        tmp[g * 8 + j] = p[j * HW];     // immediate offsets
                } else {
                    const int hi = h  + r - 1;
                    const int wi = wx + s - 1;
                    ok[g] = ((unsigned)hi < H_IMG) & ((unsigned)wi < W_IMG);
                    const int hic = min(max(hi, 0), H_IMG - 1);
                    const int wic = min(max(wi, 0), W_IMG - 1);
                    const float* p = xlane + hic * W_IMG + wic;
                    #pragma unroll
                    for (int j = 0; j < 8; ++j)
                        tmp[g * 8 + j] = p[j * HW];
                }
            } else {
                ok[g] = false;
                #pragma unroll
                for (int j = 0; j < 8; ++j) tmp[g * 8 + j] = 0.0f;
            }
        }

        v16h a;
        #pragma unroll
        for (int idx = 0; idx < 16; ++idx) {
            const float v = (!MASKED || ok[idx >> 3]) ? tmp[idx] : 0.0f;
            a[idx] = (_Float16)v;
        }

        // B fragments: lane holds K = kkbase + 8*half + s (s<8) and
        // kkbase + 16 + 8*half + (s-8) (s>=8): two 16-byte LDS slices each.
        const int kb = kkbase + half * 8;
        v16h b0, b1;
        #pragma unroll
        for (int s = 0; s < 16; ++s) {
            const int kk = kb + (s >> 3) * 16 + (s & 7);
            b0[s] = wlds[l15][kk];
            b1[s] = wlds[16 + l15][kk];
        }

        acc0 = __builtin_amdgcn_wmma_f32_16x16x32_f16(false, a, false, b0,
                                                      (short)0, acc0, false, false);
        acc1 = __builtin_amdgcn_wmma_f32_16x16x32_f16(false, a, false, b1,
                                                      (short)0, acc1, false, false);
    }
}

__global__ __launch_bounds__(256)
void conv3x3_wmma_f16(const float* __restrict__ x,
                      const float* __restrict__ w,
                      const float* __restrict__ bias,
                      float* __restrict__ out)
{
    // Weights staged as f16 in reduction order u = rs*16 + c, row-padded to 160
    // so every 32-K chunk slice is 32B-aligned; u >= 144 is zero. 10 KB LDS.
    __shared__ __align__(32) _Float16 wlds[K_OUT][RED_PAD];

    const int tid = threadIdx.x;
    for (int i = tid; i < K_OUT * RED_PAD; i += 256) {
        const int k = i / RED_PAD;
        const int u = i - k * RED_PAD;
        float v = 0.0f;
        if (u < RED) {
            const int rs = u >> 4;       // spatial tap 0..8
            const int c  = u & 15;       // input channel
            v = w[k * RED + c * 9 + rs];
        }
        wlds[k][u] = (_Float16)v;
    }
    __syncthreads();

    const int lane = tid & 31;
    const int wave = tid >> 5;
    const int half = lane >> 4;   // K-subset selector (A) / K-half (B)
    const int l15  = lane & 15;   // M row (pixel) for A, N col (channel) for B

    const int n  = blockIdx.z;
    const int h  = blockIdx.y;
    const int w0 = blockIdx.x * (WAVES * TILE_W) + wave * TILE_W;
    const int wx = w0 + l15;      // this lane's pixel column (A-matrix row)

    // Per-lane base: image n + channel block (half*8).
    const float* xlane = x + (size_t)n * C_IN * HW + ((size_t)(half * 8)) * HW;

    v8f acc0 = {};  // output channels  0..15
    v8f acc1 = {};  // output channels 16..31

    // Wave-uniform interior test; readfirstlane forces a scalar branch so the
    // WMMAs sit in straight-line code with EXEC all-ones.
    const int w0u = __builtin_amdgcn_readfirstlane(w0);
    const bool interior = (h >= 1) & (h <= H_IMG - 2) &
                          (w0u >= 1) & (w0u + TILE_W <= W_IMG - 1);
    if (interior)
        conv_tile<false>(xlane, wlds, h, wx, l15, half, acc0, acc1);
    else
        conv_tile<true >(xlane, wlds, h, wx, l15, half, acc0, acc1);

    // Reference adds bias[0] to ALL output channels.
    const float bz = bias[0];

    // C/D layout: VGPR j -> M = j (lanes 0-15) or j+8 (lanes 16-31); N = l15.
    // This lane's 8 acc elements are 8 consecutive pixels at one channel:
    // two b128 stores per accumulator.
    const int pb = half * 8;
    const size_t wcol = (size_t)(w0 + pb);

    {
        const size_t o = (((size_t)n * K_OUT + l15) * H_IMG + h) * W_IMG + wcol;
        float4 lo = make_float4(acc0[0] + bz, acc0[1] + bz, acc0[2] + bz, acc0[3] + bz);
        float4 hi = make_float4(acc0[4] + bz, acc0[5] + bz, acc0[6] + bz, acc0[7] + bz);
        *(float4*)(out + o)     = lo;
        *(float4*)(out + o + 4) = hi;
    }
    {
        const size_t o = (((size_t)n * K_OUT + 16 + l15) * H_IMG + h) * W_IMG + wcol;
        float4 lo = make_float4(acc1[0] + bz, acc1[1] + bz, acc1[2] + bz, acc1[3] + bz);
        float4 hi = make_float4(acc1[4] + bz, acc1[5] + bz, acc1[6] + bz, acc1[7] + bz);
        *(float4*)(out + o)     = lo;
        *(float4*)(out + o + 4) = hi;
    }
}

extern "C" void kernel_launch(void* const* d_in, const int* in_sizes, int n_in,
                              void* d_out, int out_size, void* d_ws, size_t ws_size,
                              hipStream_t stream) {
    const float* x  = (const float*)d_in[0];
    const float* w  = (const float*)d_in[1];
    const float* b  = (const float*)d_in[2];
    float* out      = (float*)d_out;

    dim3 grid(W_IMG / (WAVES * TILE_W), H_IMG, N_IMG);  // (2, 256, 32)
    conv3x3_wmma_f16<<<grid, 256, 0, stream>>>(x, w, b, out);
}
